// PointNetSetAbstractionMultiScaleGrouping_386547057064
// MI455X (gfx1250) — matrix operand order repaired and checked by reading, compiled, not run on hardware
//
#include <hip/hip_runtime.h>
#include <hip/hip_bf16.h>

// ---------------------------------------------------------------------------
// PointNet++ MSG set-abstraction for MI455X (gfx1250, wave32, WMMA).
//   B=8, N=8192, D=32, NPOINT=512, radii {0.1,0.2,0.4}, nsample {16,32,128}
//   MLPs: [35->32->32->64], [35->64->64->128], [35->64->96->128]
// Pipeline: prep (fold BN into f16 weights) -> FPS -> ball query (all radii,
// one scan) -> 3 fused WMMA MLP kernels (3 layers in LDS, max-pool, write).
// ---------------------------------------------------------------------------

typedef _Float16 v8h  __attribute__((ext_vector_type(8)));
typedef _Float16 v16h __attribute__((ext_vector_type(16)));
typedef float    v8f  __attribute__((ext_vector_type(8)));

static constexpr int BATCH = 8;
static constexpr int NPTS  = 8192;
static constexpr int DCH   = 32;
static constexpr int SPT   = 512;   // NPOINT
static constexpr int XS    = 136;   // LDS f16 row stride (bank-conflict pad)
static constexpr int FS    = 132;   // LDS f32 row stride

// workspace layout (bytes)
static constexpr size_t OFF_FPS = 0;                       // int[8*512]
static constexpr size_t OFF_G0  = 16384;                   // int[8*512*16]
static constexpr size_t OFF_G1  = OFF_G0 + 262144;         // int[8*512*32]
static constexpr size_t OFF_G2  = OFF_G1 + 524288;         // int[8*512*128]
static constexpr size_t OFF_W   = OFF_G2 + 2097152;        // f16[44032]
static constexpr size_t OFF_B   = OFF_W + 90112;           // f32[672]

struct ParamPtrs { const float* p[54]; };

// ---------------------------------------------------------------------------
// 1) Fold batchnorm into f16 weights + f32 bias.  grid = 9 (one per layer).
// ---------------------------------------------------------------------------
__global__ __launch_bounds__(256) void prep_weights(ParamPtrs pp,
                                                    _Float16* __restrict__ wout,
                                                    float* __restrict__ bout) {
  const int COUT[9] = {32,32,64, 64,64,128, 64,96,128};
  const int CINA[9] = {35,32,32, 35,64,64,  35,64,96};
  const int CINP[9] = {64,32,32, 64,64,64,  64,64,96};
  const int WOFF[9] = {0,2048,3072, 5120,9216,13312, 21504,25600,31744};
  const int BOFF[9] = {0,32,64, 128,192,256, 384,448,544};
  const int l = blockIdx.x;
  const float* W     = pp.p[l*6 + 0];
  const float* bias  = pp.p[l*6 + 1];
  const float* gamma = pp.p[l*6 + 2];
  const float* beta  = pp.p[l*6 + 3];
  const float* mean  = pp.p[l*6 + 4];
  const float* var   = pp.p[l*6 + 5];
  const int co = COUT[l], cia = CINA[l], cip = CINP[l];
  for (int idx = threadIdx.x; idx < co * cip; idx += blockDim.x) {
    int o = idx / cip, c = idx % cip;
    float g = gamma[o] * rsqrtf(var[o] + 1e-5f);
    float v = (c < cia) ? W[o*cia + c] * g : 0.0f;
    wout[WOFF[l] + idx] = (_Float16)v;
  }
  for (int o = threadIdx.x; o < co; o += blockDim.x) {
    float g = gamma[o] * rsqrtf(var[o] + 1e-5f);
    bout[BOFF[l] + o] = (bias[o] - mean[o]) * g + beta[o];
  }
}

// ---------------------------------------------------------------------------
// 2) Farthest point sampling. grid = B, block = 1024 (32 waves).
//    LDS: xyz (96KB) + running dist (32KB) + reduction scratch.
//    Emits fps index list and new_xyz (B,3,512) straight into d_out.
// ---------------------------------------------------------------------------
__global__ __launch_bounds__(1024) void fps_kernel(const float* __restrict__ xyz,
                                                   int* __restrict__ fpsIdx,
                                                   float* __restrict__ outXyz) {
  extern __shared__ char smem[];
  float* px   = (float*)smem;
  float* py   = px + NPTS;
  float* pz   = py + NPTS;
  float* dist = pz + NPTS;
  float* rv   = dist + NPTS;            // 32 wave maxima
  int*   ri   = (int*)(rv + 32);        // 32 wave argmax
  int*   farp = ri + 32;

  const int b = blockIdx.x, t = threadIdx.x;
  for (int i = t; i < NPTS; i += 1024) {
    px[i] = xyz[(b*3 + 0)*NPTS + i];
    py[i] = xyz[(b*3 + 1)*NPTS + i];
    pz[i] = xyz[(b*3 + 2)*NPTS + i];
    dist[i] = 1e10f;
  }
  __syncthreads();

  const int lane = t & 31, wv = t >> 5;
  int far = 0;
  for (int it = 0; it < SPT; ++it) {
    if (t == 0) {
      fpsIdx[b*SPT + it] = far;
      outXyz[(b*3 + 0)*SPT + it] = px[far];
      outXyz[(b*3 + 1)*SPT + it] = py[far];
      outXyz[(b*3 + 2)*SPT + it] = pz[far];
    }
    const float cx = px[far], cy = py[far], cz = pz[far];
    float best = -1.0f; int bi = 0;
    #pragma unroll
    for (int k = 0; k < 8; ++k) {                 // ascending i -> first-max ties
      int i = t + k*1024;
      float dx = px[i]-cx, dy = py[i]-cy, dz = pz[i]-cz;
      float d  = dx*dx + dy*dy + dz*dz;
      float nd = fminf(dist[i], d);
      dist[i] = nd;
      if (nd > best) { best = nd; bi = i; }
    }
    #pragma unroll
    for (int off = 16; off >= 1; off >>= 1) {      // wave32 shuffle argmax
      float ov = __shfl_down(best, off, 32);
      int   oi = __shfl_down(bi,   off, 32);
      if (ov > best || (ov == best && oi < bi)) { best = ov; bi = oi; }
    }
    if (lane == 0) { rv[wv] = best; ri[wv] = bi; }
    __syncthreads();
    if (wv == 0) {
      best = rv[lane]; bi = ri[lane];
      #pragma unroll
      for (int off = 16; off >= 1; off >>= 1) {
        float ov = __shfl_down(best, off, 32);
        int   oi = __shfl_down(bi,   off, 32);
        if (ov > best || (ov == best && oi < bi)) { best = ov; bi = oi; }
      }
      if (lane == 0) *farp = bi;
    }
    __syncthreads();
    far = *farp;
    __syncthreads();
  }
}

// ---------------------------------------------------------------------------
// 3) Ball query, all 3 radii in one scan. One wave per query point.
//    Sequential chunk scan + ballot prefix-compaction == reference's
//    sorted-ascending "first nsample within radius", padded with slot 0.
// ---------------------------------------------------------------------------
__global__ __launch_bounds__(256) void ball_query(const float* __restrict__ xyz,
                                                  const float* __restrict__ nxyz,
                                                  int* __restrict__ g0,
                                                  int* __restrict__ g1,
                                                  int* __restrict__ g2) {
  const int lane = threadIdx.x & 31, wv = threadIdx.x >> 5;
  const int q = blockIdx.x * 8 + wv;
  const int b = q >> 9, s = q & 511;
  const float cx = nxyz[(b*3 + 0)*SPT + s];
  const float cy = nxyz[(b*3 + 1)*SPT + s];
  const float cz = nxyz[(b*3 + 2)*SPT + s];
  int* o0 = g0 + (b*SPT + s)*16;
  int* o1 = g1 + (b*SPT + s)*32;
  int* o2 = g2 + (b*SPT + s)*128;
  const float r0 = 0.01f, r1 = 0.04f, r2 = 0.16f;   // squared radii
  int c0 = 0, c1 = 0, c2 = 0;
  const unsigned lt = (1u << lane) - 1u;

  for (int i0 = 0; i0 < NPTS; i0 += 32) {
    int i = i0 + lane;
    float dx = xyz[(b*3 + 0)*NPTS + i] - cx;
    float dy = xyz[(b*3 + 1)*NPTS + i] - cy;
    float dz = xyz[(b*3 + 2)*NPTS + i] - cz;
    float d2 = dx*dx + dy*dy + dz*dz;
    unsigned m0 = (unsigned)__ballot(d2 <= r0);
    unsigned m1 = (unsigned)__ballot(d2 <= r1);
    unsigned m2 = (unsigned)__ballot(d2 <= r2);
    if (c0 < 16 && ((m0 >> lane) & 1)) { int p = c0 + __popc(m0 & lt); if (p < 16)  o0[p] = i; }
    if (c1 < 32 && ((m1 >> lane) & 1)) { int p = c1 + __popc(m1 & lt); if (p < 32)  o1[p] = i; }
    if (c2 < 128 && ((m2 >> lane) & 1)) { int p = c2 + __popc(m2 & lt); if (p < 128) o2[p] = i; }
    c0 += __popc(m0); c1 += __popc(m1); c2 += __popc(m2);
    if (c0 >= 16 && c1 >= 32 && c2 >= 128) break;
  }
  // pad remaining slots with first in-radius index (slot 0; centroid always hits)
  int f0 = o0[0], f1 = o1[0], f2 = o2[0];
  int e0 = c0 < 16 ? c0 : 16, e1 = c1 < 32 ? c1 : 32, e2 = c2 < 128 ? c2 : 128;
  for (int j = e0 + lane; j < 16;  j += 32) o0[j] = f0;
  for (int j = e1 + lane; j < 32;  j += 32) o1[j] = f1;
  for (int j = e2 + lane; j < 128; j += 32) o2[j] = f2;
}

// ---------------------------------------------------------------------------
// 4) Fused 3-layer MLP + max-pool.  Tile = 128 rows (G groups x KS samples),
//    8 waves; wave w owns rows [16w,16w+16) -> layers run in place in LDS
//    with no inter-layer barrier (A strip fully register-resident first).
// ---------------------------------------------------------------------------
template<int CIN, int COUT, bool LAST>
__device__ __forceinline__ void mlp_layer(_Float16* __restrict__ xb,
                                          float* __restrict__ fb,
                                          const _Float16* __restrict__ W,
                                          const float* __restrict__ bias,
                                          int wave, int lane) {
  constexpr int NK = CIN / 32;
  const int mrow = wave * 16;
  const int rlane = lane & 15, khalf = lane >> 4;
  const int arow = mrow + rlane;
  v16h a[NK];
  #pragma unroll
  for (int k = 0; k < NK; ++k) {        // ISA 16-bit A 16x32 layout
    v8h lo = *(const v8h*)(xb + arow*XS + k*32 + khalf*8);
    v8h hi = *(const v8h*)(xb + arow*XS + k*32 + 16 + khalf*8);
    #pragma unroll
    for (int e = 0; e < 8; ++e) { a[k][e] = lo[e]; a[k][8+e] = hi[e]; }
  }
  #pragma unroll
  for (int nt = 0; nt < COUT/16; ++nt) {
    const int n = nt*16 + rlane;
    v8f acc = {};
    #pragma unroll
    for (int k = 0; k < NK; ++k) {      // B lane = output col, K split as A (8 + 8 at +16)
      const _Float16* wp = W + n*CIN + k*32 + khalf*8;
      v8h blo = *(const v8h*)(wp);
      v8h bhi = *(const v8h*)(wp + 16);
      v16h bf;
      #pragma unroll
      for (int e = 0; e < 8; ++e) { bf[e] = blo[e]; bf[8+e] = bhi[e]; }
      acc = __builtin_amdgcn_wmma_f32_16x16x32_f16(false, a[k], false, bf,
                                                   (short)0, acc, false, false);
    }
    const float bb = bias[n];
    #pragma unroll
    for (int v = 0; v < 8; ++v) {       // C/D layout: row = v + 8*khalf
      float y = fmaxf(acc[v] + bb, 0.0f);
      int ro = mrow + khalf*8 + v;
      if (LAST) fb[ro*FS + n] = y;
      else      xb[ro*XS + n] = (_Float16)y;
    }
  }
}

template<int KS, int G, int C1, int C2, int C3, int CHOFF>
__global__ __launch_bounds__(256) void mlp_fused(const float* __restrict__ xyz,
                                                 const float* __restrict__ feat,
                                                 const float* __restrict__ nxyz,
                                                 const int*   __restrict__ gidx,
                                                 const _Float16* __restrict__ w1,
                                                 const _Float16* __restrict__ w2,
                                                 const _Float16* __restrict__ w3,
                                                 const float* __restrict__ b1,
                                                 const float* __restrict__ b2,
                                                 const float* __restrict__ b3,
                                                 float* __restrict__ nfeat) {
  extern __shared__ char smem[];
  float*    fb = (float*)smem;                    // 128 x FS f32
  _Float16* xb = (_Float16*)(smem + 128*FS*4);    // 128 x XS f16

  const int blocksPerBatch = SPT / G;
  const int b  = blockIdx.x / blocksPerBatch;
  const int s0 = (blockIdx.x % blocksPerBatch) * G;
  const int t  = threadIdx.x;

  __builtin_prefetch(w1, 0, 1);   // global_prefetch_b8 into L2

  // build grouped input tile: cols [0,32)=features, [32,35)=xyz - centroid, pad->0
  if (t < 128) {
    const int g = t / KS, j = t % KS, s = s0 + g;
    const int idx = gidx[(b*SPT + s)*KS + j];
    _Float16* row = xb + t*XS;
    #pragma unroll 8
    for (int c = 0; c < DCH; ++c)
      row[c] = (_Float16)feat[(b*DCH + c)*NPTS + idx];
    #pragma unroll
    for (int d = 0; d < 3; ++d)
      row[DCH + d] = (_Float16)(xyz[(b*3 + d)*NPTS + idx] - nxyz[(b*3 + d)*SPT + s]);
    #pragma unroll
    for (int c = 35; c < 64; ++c) row[c] = (_Float16)0.0f;
  }
  __syncthreads();

  const int wave = t >> 5, lane = t & 31;
  mlp_layer<64, C1, false>(xb, fb, w1, b1, wave, lane);
  mlp_layer<C1, C2, false>(xb, fb, w2, b2, wave, lane);
  mlp_layer<C2, C3, true >(xb, fb, w3, b3, wave, lane);
  __syncthreads();

  // max over neighborhood, write (B, 320, 512) channel block
  for (int g = 0; g < G; ++g) {
    for (int c = t; c < C3; c += 256) {
      float m = fb[(g*KS)*FS + c];
      for (int j = 1; j < KS; ++j) m = fmaxf(m, fb[(g*KS + j)*FS + c]);
      nfeat[(b*320 + CHOFF + c)*SPT + (s0 + g)] = m;
    }
  }
}

// ---------------------------------------------------------------------------
extern "C" void kernel_launch(void* const* d_in, const int* in_sizes, int n_in,
                              void* d_out, int out_size, void* d_ws, size_t ws_size,
                              hipStream_t stream) {
  const float* xyz  = (const float*)d_in[0];
  const float* feat = (const float*)d_in[1];
  ParamPtrs pp;
  for (int i = 0; i < 54; ++i) pp.p[i] = (const float*)d_in[2 + i];

  char* ws = (char*)d_ws;
  int*      fpsIdx = (int*)(ws + OFF_FPS);
  int*      g0     = (int*)(ws + OFF_G0);
  int*      g1     = (int*)(ws + OFF_G1);
  int*      g2     = (int*)(ws + OFF_G2);
  _Float16* wbuf   = (_Float16*)(ws + OFF_W);
  float*    bbuf   = (float*)(ws + OFF_B);

  float* out   = (float*)d_out;
  float* nxyz  = out;                         // (B,3,512)
  float* nfeat = out + BATCH*3*SPT;           // (B,320,512)

  prep_weights<<<9, 256, 0, stream>>>(pp, wbuf, bbuf);

  const size_t fpsShm = (size_t)4*NPTS*4 + 32*4 + 32*4 + 16;
  fps_kernel<<<BATCH, 1024, fpsShm, stream>>>(xyz, fpsIdx, nxyz);

  ball_query<<<BATCH*SPT/8, 256, 0, stream>>>(xyz, nxyz, g0, g1, g2);

  const size_t mlpShm = (size_t)128*FS*4 + (size_t)128*XS*2;
  // weight/bias offsets per prep table
  mlp_fused<16, 8, 32, 32, 64, 0><<<BATCH*SPT/8, 256, mlpShm, stream>>>(
      xyz, feat, nxyz, g0, wbuf + 0, wbuf + 2048, wbuf + 3072,
      bbuf + 0, bbuf + 32, bbuf + 64, nfeat);
  mlp_fused<32, 4, 64, 64, 128, 64><<<BATCH*SPT/4, 256, mlpShm, stream>>>(
      xyz, feat, nxyz, g1, wbuf + 5120, wbuf + 9216, wbuf + 13312,
      bbuf + 128, bbuf + 192, bbuf + 256, nfeat);
  mlp_fused<128, 1, 64, 96, 128, 192><<<BATCH*SPT, 256, mlpShm, stream>>>(
      xyz, feat, nxyz, g2, wbuf + 21504, wbuf + 25600, wbuf + 31744,
      bbuf + 384, bbuf + 448, bbuf + 544, nfeat);
}
